// EncoderMultiHeadAttention_11665131176436
// MI455X (gfx1250) — matrix operand order repaired and checked by reading, not compiled, not run on hardware
//
#include <hip/hip_runtime.h>

typedef __attribute__((ext_vector_type(16))) _Float16 v16h;
typedef __attribute__((ext_vector_type(8)))  _Float16 v8h;
typedef __attribute__((ext_vector_type(4)))  _Float16 v4h;
typedef __attribute__((ext_vector_type(8)))  float    v8f;

constexpr int BB  = 8;
constexpr int TT  = 1024;
constexpr int DD  = 512;
constexpr int HH  = 8;
constexpr int HDC = 64;
constexpr int DD2 = 1024;

__device__ inline v16h make16(v8h lo, v8h hi) {
  return __builtin_shufflevector(lo, hi, 0,1,2,3,4,5,6,7,8,9,10,11,12,13,14,15);
}
__device__ inline v8f wmma16(v16h a, v16h b, v8f c) {
  // D = A(16x32 f16) * B(32x16 f16) + C(16x16 f32)
  return __builtin_amdgcn_wmma_f32_16x16x32_f16(false, a, false, b, (short)0, c,
                                                false, false);
}

// ---------------- conversion / init kernels ----------------
__global__ void k_cvt_f32_f16(const float* __restrict__ in,
                              _Float16* __restrict__ out, int n) {
  int i = blockIdx.x * 256 + threadIdx.x;
  if (i < n) out[i] = (_Float16)in[i];
}

__global__ void k_zero_f16(_Float16* __restrict__ p, int n) {
  int i = blockIdx.x * 256 + threadIdx.x;
  if (i < n) p[i] = (_Float16)0.f;
}

// in: [co][ci][3] f32 -> out: [co][tap][ci] f16  (B^T layout: K=ci contiguous)
__global__ void k_cvt_w(const float* __restrict__ w, _Float16* __restrict__ wh,
                        int Cin, int total) {
  int i = blockIdx.x * 256 + threadIdx.x;
  if (i >= total) return;
  int co  = i / (Cin * 3);
  int rem = i % (Cin * 3);
  int ci  = rem / 3;
  int tap = rem % 3;
  wh[((size_t)co * 3 + tap) * Cin + ci] = (_Float16)w[i];
}

// ---------------- conv1 (q/k/v projections) ----------------
// grid (BT/32=256, D/64=8, 3), block 32. Each wave: M=32 tokens, N=64 channels.
__global__ void __launch_bounds__(32)
k_conv_qkv(const _Float16* __restrict__ xh, const _Float16* __restrict__ zrow,
           const _Float16* __restrict__ whq, const _Float16* __restrict__ whk,
           const _Float16* __restrict__ whv,
           const float* __restrict__ bq, const float* __restrict__ bk,
           const float* __restrict__ bv,
           _Float16* __restrict__ qn, _Float16* __restrict__ qt,
           _Float16* __restrict__ kn, _Float16* __restrict__ kt,
           _Float16* __restrict__ vn, _Float16* __restrict__ vt) {
  const int lane = threadIdx.x;
  const int n  = lane & 15;
  const int hs = lane >> 4;
  const int mt = blockIdx.x;
  const int cg = blockIdx.y;
  const int which = blockIdx.z;
  const _Float16* wh = which == 0 ? whq : which == 1 ? whk : whv;
  const float* bias  = which == 0 ? bq  : which == 1 ? bk  : bv;
  _Float16* on = which == 0 ? qn : which == 1 ? kn : vn;
  _Float16* ot = which == 0 ? qt : which == 1 ? kt : vt;

  const int bt0 = mt * 32;
  const int b   = bt0 / TT;
  const int t0  = bt0 % TT;

  v8f acc[2][4] = {};
  for (int tap = 0; tap < 3; ++tap) {
    const _Float16* arow[2];
#pragma unroll
    for (int mi = 0; mi < 2; ++mi) {
      const int trow = t0 + mi * 16 + n + tap - 1;   // A row for this lane
      const bool ok  = (trow >= 0) && (trow < TT);
      arow[mi] = ok ? (xh + ((size_t)b * TT + trow) * DD) : zrow;  // branch-free
    }
    for (int kk = 0; kk < DD; kk += 32) {
      v16h A[2];
#pragma unroll
      for (int mi = 0; mi < 2; ++mi) {
        const v8h* ap = (const v8h*)(arow[mi] + kk + hs * 16);
        A[mi] = make16(ap[0], ap[1]);
      }
#pragma unroll
      for (int j = 0; j < 4; ++j) {
        const int co = cg * 64 + j * 16 + n;
        const v8h* bp = (const v8h*)(wh + ((size_t)co * 3 + tap) * DD + kk + hs * 16);
        v16h Bf = make16(bp[0], bp[1]);
#pragma unroll
        for (int mi = 0; mi < 2; ++mi)
          acc[mi][j] = wmma16(A[mi], Bf, acc[mi][j]);
      }
    }
  }
#pragma unroll
  for (int j = 0; j < 4; ++j) {
    const int co = cg * 64 + j * 16 + n;
    const float bvv = bias[co];
    const int h  = co >> 6;
    const int hd = co & 63;
#pragma unroll
    for (int mi = 0; mi < 2; ++mi)
#pragma unroll
      for (int r = 0; r < 8; ++r) {
        const int t = t0 + mi * 16 + r + hs * 8;
        float v = acc[mi][j][r] + bvv;
        v = v > 0.f ? v : 0.f;
        _Float16 hv = (_Float16)v;
        on[(((size_t)b * HH + h) * TT + t) * HDC + hd] = hv;   // [B,H,T,HD]
        ot[(((size_t)b * HH + h) * HDC + hd) * TT + t] = hv;   // [B,H,HD,T]
      }
  }
}

// ---------------- time-wise attention ----------------
// grid (B*H=64, T/16=64), block 32. LDS: 16x1024 f32 scores + 16x1024 f16 probs.
__global__ void __launch_bounds__(32)
k_time_attn(const _Float16* __restrict__ qn, const _Float16* __restrict__ kn,
            const _Float16* __restrict__ vt, _Float16* __restrict__ attn) {
  __shared__ float    panel[16][1024];   // 64 KB raw scores
  __shared__ _Float16 pexp[16][1024];    // 32 KB exp(s - max) as f16
  const int lane = threadIdx.x;
  const int n  = lane & 15;
  const int hs = lane >> 4;
  const int bh = blockIdx.x;
  const int b = bh >> 3, h = bh & 7;
  const int tq = blockIdx.y * 16;
  const _Float16* Qb = qn + (size_t)bh * TT * HDC;
  const _Float16* Kb = kn + (size_t)bh * TT * HDC;
  const _Float16* Vb = vt + (size_t)bh * HDC * TT;

  // phase 1: S = Q K^T / sqrt(HD).  Q fragments are loop-invariant: hoist.
  v16h A0, A1;
  {
    const v8h* ap0 = (const v8h*)(Qb + (size_t)(tq + n) * HDC + hs * 16);
    A0 = make16(ap0[0], ap0[1]);
    const v8h* ap1 = (const v8h*)(Qb + (size_t)(tq + n) * HDC + 32 + hs * 16);
    A1 = make16(ap1[0], ap1[1]);
  }
  for (int ktile = 0; ktile < 64; ++ktile) {
    v8f acc = {};
    const v8h* bp0 = (const v8h*)(Kb + (size_t)(ktile * 16 + n) * HDC + hs * 16);
    acc = wmma16(A0, make16(bp0[0], bp0[1]), acc);
    const v8h* bp1 = (const v8h*)(Kb + (size_t)(ktile * 16 + n) * HDC + 32 + hs * 16);
    acc = wmma16(A1, make16(bp1[0], bp1[1]), acc);
#pragma unroll
    for (int r = 0; r < 8; ++r)
      panel[r + hs * 8][ktile * 16 + n] = acc[r] * 0.125f;
  }
  __syncthreads();

  // phase 2: row softmax -> f16 prob panel (unnormalized; keep 1/sum in reg)
  float inv;
  {
    const int row = n;
    const int cb  = hs * 512;
    const float4* prow = (const float4*)&panel[row][cb];
    float mx = -3.0e38f;
    for (int i = 0; i < 128; ++i) {
      float4 v = prow[i];
      mx = fmaxf(mx, fmaxf(fmaxf(v.x, v.y), fmaxf(v.z, v.w)));
    }
    mx = fmaxf(mx, __shfl_xor(mx, 16, 32));
    float sum = 0.f;
    for (int i = 0; i < 128; ++i) {
      float4 v = prow[i];
      float e0 = __expf(v.x - mx), e1 = __expf(v.y - mx);
      float e2 = __expf(v.z - mx), e3 = __expf(v.w - mx);
      sum += e0 + e1 + e2 + e3;
      v4h e; e[0] = (_Float16)e0; e[1] = (_Float16)e1;
             e[2] = (_Float16)e2; e[3] = (_Float16)e3;
      *(v4h*)&pexp[row][cb + i * 4] = e;
    }
    sum += __shfl_xor(sum, 16, 32);
    inv = 1.0f / sum;           // lanes l and l+16 both hold 1/sum of row l&15
  }
  __syncthreads();

  // phase 3: O = P V  (A from f16 LDS panel, B from V^T layout [HD,T])
  v8f oacc[4] = {};
  for (int kb = 0; kb < TT; kb += 32) {
    const v8h* pr = (const v8h*)&pexp[n][kb + hs * 16];
    v16h A = make16(pr[0], pr[1]);
#pragma unroll
    for (int j = 0; j < 4; ++j) {
      const int hd = j * 16 + n;
      const v8h* bp = (const v8h*)(Vb + (size_t)hd * TT + kb + hs * 16);
      oacc[j] = wmma16(A, make16(bp[0], bp[1]), oacc[j]);
    }
  }
#pragma unroll
  for (int j = 0; j < 4; ++j) {
#pragma unroll
    for (int r = 0; r < 8; ++r) {
      const int m  = r + hs * 8;
      const int t  = tq + m;
      const int hd = j * 16 + n;
      float invm = __shfl(inv, m, 32);
      attn[((size_t)b * TT + t) * DD2 + h * 64 + hd] = (_Float16)(oacc[j][r] * invm);
    }
  }
}

// ---------------- channel-wise attention: scores + softmax ----------------
// grid (B*H=64), block 32. St = Qt Kt^T / sqrt(T), softmax rows -> At f16.
__global__ void __launch_bounds__(32)
k_chan_scores(const _Float16* __restrict__ qt, const _Float16* __restrict__ kt,
              _Float16* __restrict__ At) {
  __shared__ float sp[64][64];
  const int lane = threadIdx.x;
  const int n  = lane & 15;
  const int hs = lane >> 4;
  const int bh = blockIdx.x;
  const _Float16* Qb = qt + (size_t)bh * HDC * TT;
  const _Float16* Kb = kt + (size_t)bh * HDC * TT;

  v8f acc[16] = {};
  for (int kk = 0; kk < TT; kk += 32) {
    v16h Bf[4];
#pragma unroll
    for (int j = 0; j < 4; ++j) {
      const v8h* bp = (const v8h*)(Kb + (size_t)(j * 16 + n) * TT + kk + hs * 16);
      Bf[j] = make16(bp[0], bp[1]);
    }
#pragma unroll
    for (int i = 0; i < 4; ++i) {
      const v8h* ap = (const v8h*)(Qb + (size_t)(i * 16 + n) * TT + kk + hs * 16);
      v16h Af = make16(ap[0], ap[1]);
#pragma unroll
      for (int j = 0; j < 4; ++j)
        acc[i * 4 + j] = wmma16(Af, Bf[j], acc[i * 4 + j]);
    }
  }
#pragma unroll
  for (int i = 0; i < 4; ++i)
#pragma unroll
    for (int j = 0; j < 4; ++j)
#pragma unroll
      for (int r = 0; r < 8; ++r)
        sp[i * 16 + r + hs * 8][j * 16 + n] = acc[i * 4 + j][r] * 0.03125f;
  __syncthreads();

  for (int rr = lane * 2; rr < lane * 2 + 2; ++rr) {
    float mx = -3.0e38f;
    for (int d = 0; d < 64; ++d) mx = fmaxf(mx, sp[rr][d]);
    float sum = 0.f;
    for (int d = 0; d < 64; ++d) sum += __expf(sp[rr][d] - mx);
    const float inv = 1.0f / sum;
    for (int d = 0; d < 64; ++d)
      At[((size_t)bh * 64 + rr) * 64 + d] = (_Float16)(__expf(sp[rr][d] - mx) * inv);
  }
}

// ---------------- channel-wise attention: O = At * Vt ----------------
// grid (B*H=64, T/16=64), block 32. Writes with the reference's reshape scatter.
__global__ void __launch_bounds__(32)
k_chan_out(const _Float16* __restrict__ At, const _Float16* __restrict__ vn,
           _Float16* __restrict__ attn) {
  const int lane = threadIdx.x;
  const int n  = lane & 15;
  const int hs = lane >> 4;
  const int bh = blockIdx.x;
  const int b = bh >> 3, h = bh & 7;
  const int tt = blockIdx.y * 16;
  const _Float16* Vb = vn + (size_t)bh * TT * HDC;   // [T,HD] = (Vt)^T
  const _Float16* Ab = At + (size_t)bh * 64 * 64;

  v8f acc[4] = {};
#pragma unroll
  for (int kb = 0; kb < 64; kb += 32) {
    const v8h* bp = (const v8h*)(Vb + (size_t)(tt + n) * HDC + kb + hs * 16);
    v16h Bf = make16(bp[0], bp[1]);
#pragma unroll
    for (int i = 0; i < 4; ++i) {
      const v8h* ap = (const v8h*)(Ab + (size_t)(i * 16 + n) * 64 + kb + hs * 16);
      acc[i] = wmma16(make16(ap[0], ap[1]), Bf, acc[i]);
    }
  }
#pragma unroll
  for (int i = 0; i < 4; ++i)
#pragma unroll
    for (int r = 0; r < 8; ++r) {
      const int c = i * 16 + r + hs * 8;
      const int t = tt + n;
      const int tn = c * 16 + (t >> 6);            // reference reshape:
      const int e  = ((t & 63) << 3) | h;          // [B,HD,T,H] -> [B,T,512]
      attn[((size_t)b * TT + tn) * DD2 + 512 + e] = (_Float16)acc[i][r];
    }
}

// ---------------- conv2 (output projection) ----------------
// grid (BT/32=256, D/64=8), block 32. M=32, N=64, K = 3 taps * 1024 channels.
__global__ void __launch_bounds__(32)
k_conv2(const _Float16* __restrict__ attn, const _Float16* __restrict__ zrow,
        const _Float16* __restrict__ wh2, const float* __restrict__ b2,
        float* __restrict__ out) {
  const int lane = threadIdx.x;
  const int n  = lane & 15;
  const int hs = lane >> 4;
  const int mt = blockIdx.x;
  const int cg = blockIdx.y;
  const int bt0 = mt * 32;
  const int b   = bt0 / TT;
  const int t0  = bt0 % TT;

  v8f acc[2][4] = {};
  for (int tap = 0; tap < 3; ++tap) {
    const _Float16* arow[2];
#pragma unroll
    for (int mi = 0; mi < 2; ++mi) {
      const int trow = t0 + mi * 16 + n + tap - 1;
      const bool ok  = (trow >= 0) && (trow < TT);
      arow[mi] = ok ? (attn + ((size_t)b * TT + trow) * DD2) : zrow;
    }
    for (int kk = 0; kk < DD2; kk += 32) {
      v16h A[2];
#pragma unroll
      for (int mi = 0; mi < 2; ++mi) {
        const v8h* ap = (const v8h*)(arow[mi] + kk + hs * 16);
        A[mi] = make16(ap[0], ap[1]);
      }
#pragma unroll
      for (int j = 0; j < 4; ++j) {
        const int co = cg * 64 + j * 16 + n;
        const v8h* bp = (const v8h*)(wh2 + ((size_t)co * 3 + tap) * DD2 + kk + hs * 16);
        v16h Bf = make16(bp[0], bp[1]);
#pragma unroll
        for (int mi = 0; mi < 2; ++mi)
          acc[mi][j] = wmma16(A[mi], Bf, acc[mi][j]);
      }
    }
  }
#pragma unroll
  for (int j = 0; j < 4; ++j) {
    const int co = cg * 64 + j * 16 + n;
    const float bvv = b2[co];
#pragma unroll
    for (int mi = 0; mi < 2; ++mi)
#pragma unroll
      for (int r = 0; r < 8; ++r) {
        const int t = t0 + mi * 16 + r + hs * 8;
        float v = acc[mi][j][r] + bvv;
        v = v > 0.f ? v : 0.f;
        out[((size_t)b * TT + t) * DD + co] = v;
      }
  }
}

// ---------------- launch ----------------
extern "C" void kernel_launch(void* const* d_in, const int* in_sizes, int n_in,
                              void* d_out, int out_size, void* d_ws, size_t ws_size,
                              hipStream_t stream) {
  const float* x   = (const float*)d_in[0];
  const float* w11 = (const float*)d_in[1];
  const float* b11 = (const float*)d_in[2];
  const float* w12 = (const float*)d_in[3];
  const float* b12 = (const float*)d_in[4];
  const float* w13 = (const float*)d_in[5];
  const float* b13 = (const float*)d_in[6];
  const float* w2  = (const float*)d_in[7];
  const float* b2  = (const float*)d_in[8];
  float* out = (float*)d_out;

  _Float16* ws = (_Float16*)d_ws;
  size_t off = 0;
  auto alloc = [&](size_t nelem) { _Float16* p = ws + off; off += nelem; return p; };
  const size_t XN  = (size_t)BB * TT * DD;
  const size_t WN  = (size_t)DD * 3 * DD;
  const size_t W2N = (size_t)DD * 3 * DD2;
  const size_t QN  = (size_t)BB * HH * TT * HDC;
  _Float16* xh   = alloc(XN);
  _Float16* zrow = alloc((size_t)DD2);         // shared zero row for padding
  _Float16* whq  = alloc(WN);
  _Float16* whk  = alloc(WN);
  _Float16* whv  = alloc(WN);
  _Float16* wh2  = alloc(W2N);
  _Float16* qn = alloc(QN);
  _Float16* qt = alloc(QN);
  _Float16* kn = alloc(QN);
  _Float16* kt = alloc(QN);
  _Float16* vn = alloc(QN);
  _Float16* vt = alloc(QN);
  _Float16* attnb = alloc((size_t)BB * TT * DD2);
  _Float16* Atb   = alloc((size_t)BB * HH * 64 * 64);

  k_cvt_f32_f16<<<(int)((XN + 255) / 256), 256, 0, stream>>>(x, xh, (int)XN);
  k_zero_f16<<<(DD2 + 255) / 256, 256, 0, stream>>>(zrow, DD2);
  k_cvt_w<<<(DD * DD * 3 + 255) / 256, 256, 0, stream>>>(w11, whq, DD, DD * DD * 3);
  k_cvt_w<<<(DD * DD * 3 + 255) / 256, 256, 0, stream>>>(w12, whk, DD, DD * DD * 3);
  k_cvt_w<<<(DD * DD * 3 + 255) / 256, 256, 0, stream>>>(w13, whv, DD, DD * DD * 3);
  k_cvt_w<<<(DD * DD2 * 3 + 255) / 256, 256, 0, stream>>>(w2, wh2, DD2, DD * DD2 * 3);

  k_conv_qkv<<<dim3(BB * TT / 32, DD / 64, 3), 32, 0, stream>>>(
      xh, zrow, whq, whk, whv, b11, b12, b13, qn, qt, kn, kt, vn, vt);

  k_time_attn<<<dim3(BB * HH, TT / 16), 32, 0, stream>>>(qn, kn, vt, attnb);
  k_chan_scores<<<dim3(BB * HH), 32, 0, stream>>>(qt, kt, Atb);
  k_chan_out<<<dim3(BB * HH, TT / 16), 32, 0, stream>>>(Atb, vn, attnb);

  k_conv2<<<dim3(BB * TT / 32, DD / 64), 32, 0, stream>>>(attnb, zrow, wh2, b2, out);
}
